// SS2DBlock_53429393162316
// MI455X (gfx1250) — matrix operand order repaired and checked by reading, not compiled, not run on hardware
//
#include <hip/hip_runtime.h>
#include <cstdint>

// ---------------- problem dims ----------------
#define BSZ    4
#define DMODEL 96
#define DINNER 192
#define NDIR   4
#define DSTATE 16
#define DTRANK 6
#define CDBL   38          // DTRANK + 2*DSTATE
#define HH     32
#define WWD    32
#define LL     1024

// ---------------- vector types ----------------
typedef _Float16     v16h  __attribute__((ext_vector_type(16)));
typedef float        v8f   __attribute__((ext_vector_type(8)));
typedef unsigned int v4u   __attribute__((ext_vector_type(4)));
typedef unsigned int v8u   __attribute__((ext_vector_type(8)));

// ---------------- workspace layout (float offsets) ----------------
#define OFF_XX     0
#define OFF_Z      (OFF_XX   + BSZ*DINNER*LL)
#define OFF_XS     (OFF_Z    + BSZ*DINNER*LL)
#define OFF_XDBL   (OFF_XS   + BSZ*NDIR*DINNER*LL)
#define OFF_DELTA  (OFF_XDBL + BSZ*NDIR*CDBL*LL)
#define OFF_YS     (OFF_DELTA+ BSZ*NDIR*DINNER*LL)
#define OFF_Y      (OFF_YS   + BSZ*NDIR*DINNER*LL)
#define OFF_PART   (OFF_Y    + BSZ*DINNER*LL)
#define OFF_MST    (OFF_PART + BSZ*768*2)

// ---------------- helpers ----------------
// Fast SiLU: x * v_rcp_f32(1 + exp(-x)) -- avoids the IEEE divide expansion
// (v_div_scale/v_div_fmas chain) the precise '/' lowers to.
__device__ __forceinline__ float silu(float v) {
    return v * __builtin_amdgcn_rcpf(1.0f + __expf(-v));
}

// A-fragment K index, contiguous order k = 16*g + i (matches B fills):
// lane group g (lane>>4), element i in v16h.
__device__ __forceinline__ int akidx(int g, int i) {
    return g*16 + i;
}

__device__ __forceinline__ v8f wmma_f16(v16h a, v16h b, v8f c) {
    // 8 args: (neg_a, A, neg_b, B, c_mod, C, reuse_a, reuse_b)
    return __builtin_amdgcn_wmma_f32_16x16x32_f16(false, a, false, b, (short)0, c, false, false);
}

// ============================================================
// K1: in_proj GEMM (384x96 @ 96x1024 per b) + split + silu(z)
// ============================================================
__global__ void k_inproj(const float* __restrict__ x, const float* __restrict__ w,
                         float* __restrict__ xx, float* __restrict__ zz) {
    const int wave = blockIdx.x * 8 + (threadIdx.x >> 5);
    const int lane = threadIdx.x & 31;
    const int g = lane >> 4, ln = lane & 15;
    const int tn = wave & 63;
    const int tm = (wave >> 6) % 24;
    const int b  = wave / (64 * 24);
    const int col  = tn * 16 + ln;
    const int rowA = tm * 16 + ln;
    v8f acc = {};
#pragma unroll
    for (int kt = 0; kt < 3; ++kt) {
        v16h va, vb;
#pragma unroll
        for (int i = 0; i < 16; ++i) {
            va[i] = (_Float16)w[rowA * DMODEL + kt*32 + akidx(g, i)];
            vb[i] = (_Float16)x[(b * DMODEL + kt*32 + g*16 + i) * LL + col];
        }
        acc = wmma_f16(va, vb, acc);
    }
#pragma unroll
    for (int r = 0; r < 8; ++r) {
        const int o = tm*16 + g*8 + r;
        const float v = acc[r];
        if (o < DINNER) xx[(b*DINNER + o) * LL + col] = v;
        else            zz[(b*DINNER + (o - DINNER)) * LL + col] = silu(v);
    }
}

// ============================================================
// K2: fused depthwise 3x3 conv + SiLU + 4-direction cross-scan
// ============================================================
__global__ void k_scanprep(const float* __restrict__ xx, const float* __restrict__ cw,
                           const float* __restrict__ cb, float* __restrict__ xs) {
    const int tid = blockIdx.x * 256 + threadIdx.x;
    const int l    = tid & (LL - 1);
    const int d    = (tid >> 10) % DINNER;
    const int kdir = (tid / (LL * DINNER)) & 3;
    const int b    = tid / (LL * DINNER * NDIR);
    int hx, wx;
    if (kdir == 0)      { int j = l >> 5, i = l & 31; hx = (j & 1) ? 31 - i : i; wx = j; }
    else if (kdir == 1) { int j = l >> 5, i = l & 31; int s = (j & 1) ? 31 - i : i; hx = 31 - s; wx = 31 - j; }
    else if (kdir == 2) { int i = l >> 5, j = l & 31; hx = i; wx = (i & 1) ? 31 - j : j; }
    else                { int i = l >> 5, j = l & 31; int s = (i & 1) ? 31 - j : j; hx = 31 - i; wx = 31 - s; }
    float acc = cb[d];
    const float* xp = xx + (b * DINNER + d) * LL;
#pragma unroll
    for (int dh = -1; dh <= 1; ++dh) {
        const int h2 = hx + dh; if (h2 < 0 || h2 >= HH) continue;
#pragma unroll
        for (int dw = -1; dw <= 1; ++dw) {
            const int w2 = wx + dw; if (w2 < 0 || w2 >= WWD) continue;
            acc += xp[h2 * WWD + w2] * cw[d*9 + (dh+1)*3 + (dw+1)];
        }
    }
    xs[((b * NDIR + kdir) * DINNER + d) * LL + l] = silu(acc);
}

// ============================================================
// K3: x_proj GEMM (38x192 @ 192x1024 per (b,k)) -> x_dbl
// ============================================================
__global__ void k_xproj(const float* __restrict__ xs, const float* __restrict__ xpw,
                        float* __restrict__ xdbl) {
    const int wave = blockIdx.x * 8 + (threadIdx.x >> 5);
    const int lane = threadIdx.x & 31;
    const int g = lane >> 4, ln = lane & 15;
    const int tn = wave & 63;
    const int tm = (wave >> 6) % 3;
    const int kk = (wave / (64 * 3)) & 3;
    const int b  = wave / (64 * 3 * 4);
    const int col  = tn * 16 + ln;
    const int rowA = tm * 16 + ln;
    v8f acc = {};
#pragma unroll
    for (int kt = 0; kt < 6; ++kt) {
        v16h va, vb;
#pragma unroll
        for (int i = 0; i < 16; ++i) {
            va[i] = (rowA < CDBL) ? (_Float16)xpw[(kk * CDBL + rowA) * DINNER + kt*32 + akidx(g, i)]
                                  : (_Float16)0.0f;
            vb[i] = (_Float16)xs[((b * NDIR + kk) * DINNER + kt*32 + g*16 + i) * LL + col];
        }
        acc = wmma_f16(va, vb, acc);
    }
#pragma unroll
    for (int r = 0; r < 8; ++r) {
        const int c = tm*16 + g*8 + r;
        if (c < CDBL) xdbl[((b * NDIR + kk) * CDBL + c) * LL + col] = acc[r];
    }
}

// ============================================================
// K4: dt projection (rank 6) + bias + softplus -> delta
// ============================================================
__global__ void k_delta(const float* __restrict__ xdbl, const float* __restrict__ dtw,
                        const float* __restrict__ dtb, float* __restrict__ delta) {
    const int tid = blockIdx.x * 256 + threadIdx.x;
    const int l = tid & (LL - 1);
    const int d = (tid >> 10) % DINNER;
    const int k = (tid / (LL * DINNER)) & 3;
    const int b = tid / (LL * DINNER * NDIR);
    float acc = dtb[k * DINNER + d];
    const float* xp = xdbl + ((b * NDIR + k) * CDBL) * LL + l;
    const float* wp = dtw + (k * DINNER + d) * DTRANK;
#pragma unroll
    for (int r = 0; r < DTRANK; ++r) acc += xp[r * LL] * wp[r];
    delta[((b * NDIR + k) * DINNER + d) * LL + l] = (acc > 20.0f) ? acc : log1pf(__expf(acc));
}

// ============================================================
// K5: selective scan. One block per (b,k); d = thread (192).
// B/C rows (32 x 64-step chunk) staged into LDS via the TDM
// (tensor_load_to_lds + s_wait_tensorcnt), issued by wave 0.
// ============================================================
__global__ void k_scan(const float* __restrict__ xs, const float* __restrict__ delta,
                       const float* __restrict__ xdbl, const float* __restrict__ alog,
                       const float* __restrict__ dsp, float* __restrict__ ys) {
    __shared__ float bc[32 * 64];      // rows c = DTRANK..DTRANK+31, 64 timesteps
    const int b = blockIdx.x >> 2, k = blockIdx.x & 3;
    const int d = threadIdx.x;
    float h[DSTATE], An[DSTATE];
#pragma unroll
    for (int n = 0; n < DSTATE; ++n) {
        h[n]  = 0.0f;
        An[n] = -__expf(alog[(k * DINNER + d) * DSTATE + n]);
    }
    const float Dp = dsp[k * DINNER + d];
    const int chanBase = ((b * NDIR + k) * DINNER + d) * LL;
    const int bcBase   = ((b * NDIR + k) * CDBL + DTRANK) * LL;
    const unsigned ldsAddr = (unsigned)(uintptr_t)(&bc[0]);

    for (int cc = 0; cc < LL / 64; ++cc) {
        __syncthreads();                               // previous chunk fully consumed
        if ((threadIdx.x >> 5) == 0) {
            const unsigned long long ga =
                (unsigned long long)(uintptr_t)(xdbl + bcBase + cc * 64);
            v4u g0; v8u g1;
            // D# group 0: count=1, lds_addr, 57-bit global addr, type=2
            g0.x = 1u;
            g0.y = ldsAddr;
            g0.z = (unsigned)(ga & 0xffffffffu);
            g0.w = (unsigned)((ga >> 32) & 0x1ffffffu) | (2u << 30);
            // D# group 1: data_size=4B; tensor 1024 x 32; tile 64 x 32; stride0=1024
            g1.s0 = (2u << 16);          // data_size = 2 (4 bytes)
            g1.s1 = (1024u << 16);       // tensor_dim0[15:0]  @ bits 63:48
            g1.s2 = (32u   << 16);       // tensor_dim1[15:0]  @ bits 95:80 (dim0 hi16 = 0)
            g1.s3 = (64u   << 16);       // tile_dim0          @ bits 127:112 (dim1 hi16 = 0)
            g1.s4 = 32u;                 // tile_dim1 = 32, tile_dim2 = 0
            g1.s5 = 1024u;               // tensor_dim0_stride low32
            g1.s6 = 0u;
            g1.s7 = 0u;
            asm volatile("tensor_load_to_lds %0, %1" :: "s"(g0), "s"(g1) : "memory");
        }
        __builtin_amdgcn_s_wait_tensorcnt(0);          // no-op for non-issuing waves
        __syncthreads();

        for (int t = 0; t < 64; ++t) {
            const int l = cc * 64 + t;
            const float u  = xs[chanBase + l];
            const float dl = delta[chanBase + l];
            const float du = dl * u;
            float y = 0.0f;
#pragma unroll
            for (int n = 0; n < DSTATE; ++n) {
                const float dA = __expf(dl * An[n]);
                h[n] = dA * h[n] + du * bc[n * 64 + t];
                y   += h[n] * bc[(DSTATE + n) * 64 + t];
            }
            ys[chanBase + l] = y + Dp * u;
        }
    }
}

// ============================================================
// K6: cross-merge (sum of 4 inverse snake permutations)
//     + per-block partial sums for normalization
// ============================================================
__global__ void k_merge(const float* __restrict__ ys, float* __restrict__ y,
                        float* __restrict__ part) {
    __shared__ float ss[256], sq[256];
    const int tid = threadIdx.x;
    const int b   = blockIdx.x / 768;
    const int idx = (blockIdx.x % 768) * 256 + tid;    // [0, 192*1024)
    const int c  = idx >> 10;
    const int hw = idx & 1023;
    const int hx = hw >> 5, wx = hw & 31;
    const int i0 = (wx & 1) ? 31 - hx : hx;
    const int j2 = (hx & 1) ? 31 - wx : wx;
    const int l0 = wx * 32 + i0;
    const int l1 = (31 - wx) * 32 + i0;
    const int l2 = hx * 32 + j2;
    const int l3 = (31 - hx) * 32 + j2;
    const int base = b * (NDIR * DINNER * LL) + c * LL;
    const float v = ys[base + l0]
                  + ys[base + 1 * DINNER * LL + l1]
                  + ys[base + 2 * DINNER * LL + l2]
                  + ys[base + 3 * DINNER * LL + l3];
    y[(b * DINNER + c) * LL + hw] = v;
    ss[tid] = v; sq[tid] = v * v;
    __syncthreads();
    for (int s = 128; s > 0; s >>= 1) {
        if (tid < s) { ss[tid] += ss[tid + s]; sq[tid] += sq[tid + s]; }
        __syncthreads();
    }
    if (tid == 0) { part[blockIdx.x * 2] = ss[0]; part[blockIdx.x * 2 + 1] = sq[0]; }
}

// ============================================================
// K7: reduce partials -> mu, rsig per batch (deterministic)
// ============================================================
__global__ void k_stats(const float* __restrict__ part, float* __restrict__ mstat) {
    __shared__ float ss[256], sq[256];
    const int b = blockIdx.x, tid = threadIdx.x;
    float s = 0.0f, q = 0.0f;
    for (int i = tid; i < 768; i += 256) {
        s += part[(b * 768 + i) * 2];
        q += part[(b * 768 + i) * 2 + 1];
    }
    ss[tid] = s; sq[tid] = q;
    __syncthreads();
    for (int st = 128; st > 0; st >>= 1) {
        if (tid < st) { ss[tid] += ss[tid + st]; sq[tid] += sq[tid + st]; }
        __syncthreads();
    }
    if (tid == 0) {
        const float N  = (float)(DINNER * LL);
        const float mu = ss[0] / N;
        const float var = sq[0] / N - mu * mu;
        mstat[b * 2]     = mu;
        mstat[b * 2 + 1] = rsqrtf(var + 1e-6f);
    }
}

// ============================================================
// K8: out_proj GEMM (96x192 @ 192x1024 per b), with the
// norm*gamma+beta and *silu(z) fused into the B-fragment load
// ============================================================
__global__ void k_outproj(const float* __restrict__ y, const float* __restrict__ zz,
                          const float* __restrict__ mstat, const float* __restrict__ gam,
                          const float* __restrict__ bet, const float* __restrict__ wo,
                          float* __restrict__ out) {
    const int wave = blockIdx.x * 8 + (threadIdx.x >> 5);
    const int lane = threadIdx.x & 31;
    const int g = lane >> 4, ln = lane & 15;
    const int tn = wave & 63;
    const int tm = (wave >> 6) % 6;
    const int b  = wave / (64 * 6);
    const int col  = tn * 16 + ln;
    const int rowA = tm * 16 + ln;
    const float mu = mstat[b * 2], rs = mstat[b * 2 + 1];
    v8f acc = {};
#pragma unroll
    for (int kt = 0; kt < 6; ++kt) {
        v16h va, vb;
#pragma unroll
        for (int i = 0; i < 16; ++i) {
            va[i] = (_Float16)wo[rowA * DINNER + kt*32 + akidx(g, i)];
            const int c  = kt*32 + g*16 + i;
            const float yv = y[(b * DINNER + c) * LL + col];
            const float nv = (yv - mu) * rs * gam[c] + bet[c];
            vb[i] = (_Float16)(nv * zz[(b * DINNER + c) * LL + col]);
        }
        acc = wmma_f16(va, vb, acc);
    }
#pragma unroll
    for (int r = 0; r < 8; ++r) {
        const int o = tm*16 + g*8 + r;
        out[(b * DMODEL + o) * LL + col] = acc[r];
    }
}

// ============================================================
extern "C" void kernel_launch(void* const* d_in, const int* in_sizes, int n_in,
                              void* d_out, int out_size, void* d_ws, size_t ws_size,
                              hipStream_t stream) {
    const float* x    = (const float*)d_in[0];
    const float* ipw  = (const float*)d_in[1];
    const float* cw   = (const float*)d_in[2];
    const float* cb   = (const float*)d_in[3];
    const float* xpw  = (const float*)d_in[4];
    const float* dtw  = (const float*)d_in[5];
    const float* dtb  = (const float*)d_in[6];
    const float* alog = (const float*)d_in[7];
    const float* dsp  = (const float*)d_in[8];
    const float* gam  = (const float*)d_in[9];
    const float* bet  = (const float*)d_in[10];
    const float* opw  = (const float*)d_in[11];

    float* ws    = (float*)d_ws;
    float* xx    = ws + OFF_XX;
    float* zz    = ws + OFF_Z;
    float* xs    = ws + OFF_XS;
    float* xdbl  = ws + OFF_XDBL;
    float* delta = ws + OFF_DELTA;
    float* ysb   = ws + OFF_YS;
    float* yb    = ws + OFF_Y;
    float* part  = ws + OFF_PART;
    float* mstat = ws + OFF_MST;

    (void)in_sizes; (void)n_in; (void)out_size; (void)ws_size;

    // in_proj: B*24*64 = 6144 wave-tiles, 8 waves/block
    k_inproj<<<768, 256, 0, stream>>>(x, ipw, xx, zz);
    // conv + silu + cross_scan: 4*4*192*1024 threads
    k_scanprep<<<(BSZ * NDIR * DINNER * LL) / 256, 256, 0, stream>>>(xx, cw, cb, xs);
    // x_proj: B*K*3*64 = 3072 wave-tiles
    k_xproj<<<384, 256, 0, stream>>>(xs, xpw, xdbl);
    // dt projection + softplus
    k_delta<<<(BSZ * NDIR * DINNER * LL) / 256, 256, 0, stream>>>(xdbl, dtw, dtb, delta);
    // selective scan: one block per (b,k)
    k_scan<<<16, 192, 0, stream>>>(xs, delta, xdbl, alog, dsp, ysb);
    // merge + partial stats
    k_merge<<<BSZ * 768, 256, 0, stream>>>(ysb, yb, part);
    // finalize mu / rsig per batch
    k_stats<<<BSZ, 256, 0, stream>>>(part, mstat);
    // out_proj with fused norm * z
    k_outproj<<<192, 256, 0, stream>>>(yb, zz, mstat, gam, bet, opw, (float*)d_out);
}
